// QLinearSim_15195594293639
// MI455X (gfx1250) — matrix operand order repaired and checked by reading, not compiled, not run on hardware
//
#include <hip/hip_runtime.h>
#include <stdint.h>

typedef __attribute__((ext_vector_type(16))) _Float16 v16h;
typedef __attribute__((ext_vector_type(8)))  _Float16 v8h;
typedef __attribute__((ext_vector_type(8)))  float    v8f;
typedef __attribute__((ext_vector_type(4)))  float    v4f;
typedef __attribute__((ext_vector_type(4)))  int      v4i;

#define M_DIM 8192   // B * S
#define N_DIM 4096   // D_OUT
#define K_DIM 4096   // D_IN
#define BM 128
#define BN 128
#define BK 64
#define PADK 72      // 64 K-halfs + 8 pad halfs -> 144B row stride (36 dwords, conflict-free)
#define NCHUNK (K_DIM / BK)

__global__ __launch_bounds__(256) void qlinear_q4_wmma(
    const float* __restrict__ X,      // [M, K] fp32
    const int8_t* __restrict__ Q,     // [N, K] int4-in-int8
    const float* __restrict__ scale,  // [N]
    const float* __restrict__ bias,   // [N]
    float* __restrict__ out)          // [M, N] fp32
{
  __shared__ _Float16 Xs[2][BM * PADK];   // f16 activations tile
  __shared__ _Float16 Qs[2][BN * PADK];   // f16 (exact) weight tile

  const int t    = threadIdx.x;
  const int lane = t & 31;
  const int wave = t >> 5;        // 8 waves
  const int l16  = lane & 15;
  const int hi   = lane >> 4;     // lane half: selects K sub-group
  const int wm   = wave >> 2;     // 0..1 -> 64 M-rows each
  const int wn   = wave & 3;      // 0..3 -> 32 N-cols each

  const int bm = blockIdx.y * BM;
  const int bn = blockIdx.x * BN;

  // Staging assignment: 2 threads per tile row, each handles 32 K-elements.
  const int sr = t >> 1;          // row 0..127
  const int sh = (t & 1) * 32;    // K offset within chunk

  const float*  Xg = X + (size_t)(bm + sr) * K_DIM + sh;
  const int8_t* Qg = Q + (size_t)(bn + sr) * K_DIM + sh;

  // Convert 32 fp32 + 32 int8 held in registers to f16 and store into LDS buf.
  auto cvtStore = [&](const v4f* xv, const v4i* qv, int buf) {
#pragma unroll
    for (int j = 0; j < 4; ++j) {
      v8h o;
#pragma unroll
      for (int e = 0; e < 8; ++e) o[e] = (_Float16)xv[2 * j + (e >> 2)][e & 3];
      *(v8h*)&Xs[buf][sr * PADK + sh + 8 * j] = o;
    }
#pragma unroll
    for (int j = 0; j < 4; ++j) {
      v8h o;
#pragma unroll
      for (int e = 0; e < 8; ++e) {
        const int word = 2 * j + (e >> 2);
        const int wv   = qv[word >> 2][word & 3];
        o[e] = (_Float16)(float)(int8_t)(uint32_t(wv) >> ((e & 3) * 8));
      }
      *(v8h*)&Qs[buf][sr * PADK + sh + 8 * j] = o;
    }
  };

  const v8f vzero = {0.f, 0.f, 0.f, 0.f, 0.f, 0.f, 0.f, 0.f};
  v8f acc[4][2];
#pragma unroll
  for (int mt = 0; mt < 4; ++mt)
#pragma unroll
    for (int nt = 0; nt < 2; ++nt) acc[mt][nt] = vzero;

  // ---- prologue: stage chunk 0 into buffer 0 ----
  {
    const v4f* xp = (const v4f*)Xg;
    const v4i* qp = (const v4i*)Qg;
    v4f xv[8];
    v4i qv[2];
#pragma unroll
    for (int i = 0; i < 8; ++i) xv[i] = xp[i];
    qv[0] = qp[0];
    qv[1] = qp[1];
    cvtStore(xv, qv, 0);
  }
  __syncthreads();

  for (int c = 0; c < NCHUNK; ++c) {
    const int cur = c & 1;

    // Issue global loads for the next chunk early (hide under WMMA).
    v4f xv[8];
    v4i qv[2];
    if (c + 1 < NCHUNK) {
      const v4f* xp = (const v4f*)(Xg + (size_t)(c + 1) * BK);
      const v4i* qp = (const v4i*)(Qg + (size_t)(c + 1) * BK);
#pragma unroll
      for (int i = 0; i < 8; ++i) xv[i] = xp[i];
      qv[0] = qp[0];
      qv[1] = qp[1];
    }
    if (c + 2 < NCHUNK) {  // global_prefetch_b8 for chunk after next
      __builtin_prefetch(Xg + (size_t)(c + 2) * BK, 0, 3);
      __builtin_prefetch(Qg + (size_t)(c + 2) * BK, 0, 3);
    }

    // ---- compute on buffer `cur`: 2 k-steps of 32, 8 WMMAs each ----
#pragma unroll
    for (int ks = 0; ks < BK; ks += 32) {
      v16h a[4], b[2];
#pragma unroll
      for (int mt = 0; mt < 4; ++mt) {
        // A 16x32 f16 layout: lane row = l16; lanes 0-15 K {0-7,16-23}, 16-31 K {8-15,24-31}
        const _Float16* p =
            &Xs[cur][(wm * 64 + mt * 16 + l16) * PADK + ks + hi * 8];
        v8h lo = *(const v8h*)p;
        v8h hh = *(const v8h*)(p + 16);
        a[mt] = __builtin_shufflevector(lo, hh, 0, 1, 2, 3, 4, 5, 6, 7, 8, 9,
                                        10, 11, 12, 13, 14, 15);
      }
#pragma unroll
      for (int nt = 0; nt < 2; ++nt) {
        // B 32x16 f16 layout: lane col = l16; lanes 0-15 K {0-7,16-23}, 16-31 K {8-15,24-31}
        const _Float16* p =
            &Qs[cur][(wn * 32 + nt * 16 + l16) * PADK + ks + hi * 8];
        v8h lo = *(const v8h*)p;
        v8h hh = *(const v8h*)(p + 16);
        b[nt] = __builtin_shufflevector(lo, hh, 0, 1, 2, 3, 4, 5, 6, 7, 8, 9,
                                        10, 11, 12, 13, 14, 15);
      }
#pragma unroll
      for (int mt = 0; mt < 4; ++mt)
#pragma unroll
        for (int nt = 0; nt < 2; ++nt)
          acc[mt][nt] = __builtin_amdgcn_wmma_f32_16x16x32_f16(
              false, a[mt], false, b[nt], (short)0, acc[mt][nt], false, false);
    }

    // ---- stage next chunk into the other buffer ----
    if (c + 1 < NCHUNK) cvtStore(xv, qv, cur ^ 1);
    __syncthreads();
  }

  // ---- epilogue: apply per-channel scale + bias in fp32, store ----
#pragma unroll
  for (int nt = 0; nt < 2; ++nt) {
    const int n   = bn + wn * 32 + nt * 16 + l16;
    const float s = scale[n];
    const float bb = bias[n];
#pragma unroll
    for (int mt = 0; mt < 4; ++mt) {
#pragma unroll
      for (int r = 0; r < 8; ++r) {
        // C/D layout: VGPR r -> row r (lanes 0-15) / row 8+r (lanes 16-31), col = l16
        const int m = bm + wm * 64 + mt * 16 + hi * 8 + r;
        out[(size_t)m * N_DIM + n] = acc[mt][nt][r] * s + bb;
      }
    }
  }
}

extern "C" void kernel_launch(void* const* d_in, const int* in_sizes, int n_in,
                              void* d_out, int out_size, void* d_ws,
                              size_t ws_size, hipStream_t stream) {
  (void)in_sizes; (void)n_in; (void)out_size; (void)d_ws; (void)ws_size;
  const float*  x     = (const float*)d_in[0];
  const int8_t* qw    = (const int8_t*)d_in[1];
  const float*  scale = (const float*)d_in[2];
  const float*  bias  = (const float*)d_in[3];
  float*        out   = (float*)d_out;

  dim3 grid(N_DIM / BN, M_DIM / BM);  // (32, 64) workgroups
  qlinear_q4_wmma<<<grid, 256, 0, stream>>>(x, qw, scale, bias, out);
}
